// TransformerBlock_61117384622703
// MI455X (gfx1250) — matrix operand order repaired and checked by reading, not compiled, not run on hardware
//
#include <hip/hip_runtime.h>
#include <math.h>

// ---------------- problem constants ----------------
#define BQ   8
#define SEQ  1024
#define DIM  1024
#define NH   16
#define DKH  64
#define DFFN 4096

typedef __attribute__((ext_vector_type(16))) __bf16 v16bf;
typedef __attribute__((ext_vector_type(8)))  __bf16 v8bf;
typedef __attribute__((ext_vector_type(8)))  float  v8f;

// ---------------- helpers ----------------
__device__ inline v16bf cat8(v8bf lo, v8bf hi) {
  return __builtin_shufflevector(lo, hi, 0,1,2,3,4,5,6,7,8,9,10,11,12,13,14,15);
}

// A-fragment (16xK=32 bf16), ISA 7.12.2: lane r=l%16 holds row r;
// lanes<16 hold K {0..7,16..23}, lanes>=16 hold K {8..15,24..31}.
__device__ inline v16bf frag_a(const __bf16* base, int stride, int lane) {
  const int r = lane & 15, kh = lane >> 4;
  const __bf16* p = base + r * stride + kh * 8;
  return cat8(*(const v8bf*)p, *(const v8bf*)(p + 16));
}

// B-fragment (K=32 x 16 bf16): lane c=l%16 holds column c;
// lanes<16 hold K {0..7,16..23}, lanes>=16 hold K {8..15,24..31}.
// LDS tile is stored [n][k] (k contiguous), so this is 2x16B loads.
__device__ inline v16bf frag_b(const __bf16* base, int stride, int koff, int lane) {
  const int c = lane & 15, kh = lane >> 4;
  const __bf16* p = base + c * stride + koff + kh * 8;
  return cat8(*(const v8bf*)p, *(const v8bf*)(p + 16));
}

__device__ inline v8f wmma_bf16(v16bf a, v16bf b, v8f c) {
  return __builtin_amdgcn_wmma_f32_16x16x32_bf16(
      /*neg_a=*/false, a, /*neg_b=*/false, b,
      /*c_mod=*/(short)0, c, /*reuse_a=*/false, /*reuse_b=*/false);
}

// Async global->LDS copy, 16 bytes per lane (tracked by ASYNCcnt).
__device__ inline void async_ld16(unsigned lds_off, const __bf16* g) {
  unsigned long long ga = (unsigned long long)(size_t)g;
  asm volatile("global_load_async_to_lds_b128 %0, %1, off"
               :: "v"(lds_off), "v"(ga) : "memory");
}
__device__ inline void wait_async0() {
  asm volatile("s_wait_asynccnt 0x0" ::: "memory");
}

__device__ inline float rmax16(float v) {
  #pragma unroll
  for (int o = 1; o < 16; o <<= 1) v = fmaxf(v, __shfl_xor(v, o, 32));
  return v;
}
__device__ inline float rsum16(float v) {
  #pragma unroll
  for (int o = 1; o < 16; o <<= 1) v += __shfl_xor(v, o, 32);
  return v;
}

__device__ inline float gelu_f(float x) {
  const float c = 0.7978845608028654f; // sqrt(2/pi)
  return 0.5f * x * (1.f + tanhf(c * (x + 0.044715f * x * x * x)));
}

// ---------------- f32 -> bf16 convert ----------------
__global__ __launch_bounds__(256)
void cvt_f32_bf16(const float* __restrict__ in, __bf16* __restrict__ out, int n) {
  int i = blockIdx.x * blockDim.x + threadIdx.x;
  int stride = gridDim.x * blockDim.x;
  for (; i < n; i += stride) out[i] = (__bf16)in[i];
}

// ---------------- LayerNorm (ddof=1, eps on std) -> bf16 ----------------
__global__ __launch_bounds__(256)
void layernorm_to_bf16(const float* __restrict__ x, const float* __restrict__ g,
                       const float* __restrict__ bta, __bf16* __restrict__ out) {
  __shared__ float red[256];
  const int row = blockIdx.x, tid = threadIdx.x;
  const float* xr = x + (size_t)row * DIM;

  float s = 0.f;
  for (int i = tid; i < DIM; i += 256) s += xr[i];
  red[tid] = s; __syncthreads();
  for (int o = 128; o > 0; o >>= 1) { if (tid < o) red[tid] += red[tid + o]; __syncthreads(); }
  const float mean = red[0] / (float)DIM;
  __syncthreads();

  float v = 0.f;
  for (int i = tid; i < DIM; i += 256) { float d = xr[i] - mean; v += d * d; }
  red[tid] = v; __syncthreads();
  for (int o = 128; o > 0; o >>= 1) { if (tid < o) red[tid] += red[tid + o]; __syncthreads(); }
  const float inv = 1.f / (sqrtf(red[0] / (float)(DIM - 1)) + 1e-5f);

  for (int i = tid; i < DIM; i += 256)
    out[(size_t)row * DIM + i] = (__bf16)(g[i] * (xr[i] - mean) * inv + bta[i]);
}

// ---------------- WMMA GEMM: out = epilogue(A[M,K] @ W[N,K]^T + bias) ----------------
// Double-buffered LDS tiles filled by async global->LDS copies (ASYNCcnt),
// prefetch of slice k+1 overlaps WMMA on slice k.
enum { EP_HEADS = 1, EP_GELU = 2, EP_RES = 3 };

template <int MODE>
__global__ __launch_bounds__(256)
void gemm_bf16_wmma(const __bf16* __restrict__ A, const __bf16* __restrict__ W,
                    const float* __restrict__ bias, const float* __restrict__ res,
                    float* __restrict__ outF, __bf16* __restrict__ outB,
                    int M, int N, int K) {
  __shared__ __bf16 sA[2][128 * 40];  // [m][k], stride 40 (pad breaks bank conflicts)
  __shared__ __bf16 sB[2][128 * 40];  // [n][k]

  const int tid = threadIdx.x;
  const int lane = tid & 31, wave = tid >> 5;
  const int wm = wave >> 2, wn = wave & 3;       // 2x4 wave grid
  const int m0 = blockIdx.y * 128, n0 = blockIdx.x * 128;

  v8f zero = {};
  v8f acc[4][2];
  #pragma unroll
  for (int i = 0; i < 4; ++i)
    #pragma unroll
    for (int j = 0; j < 2; ++j) acc[i][j] = zero;

  const int lrow = tid >> 2;          // 0..63
  const int lcol = (tid & 3) << 3;    // 0,8,16,24

  auto prefetch = [&](int k0, int buf) {
    #pragma unroll
    for (int p = 0; p < 2; ++p) {
      const int row = lrow + p * 64;
      async_ld16((unsigned)(size_t)&sA[buf][row * 40 + lcol],
                 &A[(size_t)(m0 + row) * K + k0 + lcol]);
      async_ld16((unsigned)(size_t)&sB[buf][row * 40 + lcol],
                 &W[(size_t)(n0 + row) * K + k0 + lcol]);
    }
  };

  prefetch(0, 0);

  int buf = 0;
  for (int k0 = 0; k0 < K; k0 += 32, buf ^= 1) {
    wait_async0();        // own async writes landed...
    __syncthreads();      // ...and everyone else's too; prior reads of buf^1 done
    if (k0 + 32 < K) prefetch(k0 + 32, buf ^ 1);  // overlaps compute below

    v16bf bfr[2];
    #pragma unroll
    for (int sn = 0; sn < 2; ++sn)
      bfr[sn] = frag_b(&sB[buf][(wn * 32 + sn * 16) * 40], 40, 0, lane);
    #pragma unroll
    for (int sm = 0; sm < 4; ++sm) {
      v16bf a = frag_a(&sA[buf][(wm * 64 + sm * 16) * 40], 40, lane);
      #pragma unroll
      for (int sn = 0; sn < 2; ++sn)
        acc[sm][sn] = wmma_bf16(a, bfr[sn], acc[sm][sn]);
    }
  }

  // Epilogue: C layout — VGPR j: lanes 0-15 row j, lanes 16-31 row j+8; col = lane%16.
  const int r = lane & 15, kh = lane >> 4;
  #pragma unroll
  for (int sm = 0; sm < 4; ++sm) {
    #pragma unroll
    for (int sn = 0; sn < 2; ++sn) {
      #pragma unroll
      for (int j = 0; j < 8; ++j) {
        const int row = m0 + wm * 64 + sm * 16 + j + kh * 8;
        const int col = n0 + wn * 32 + sn * 16 + r;
        float v = acc[sm][sn][j] + bias[col];
        if constexpr (MODE == EP_HEADS) {
          // write [B,H,S,DK] for attention
          const int bi = row >> 10, s = row & (SEQ - 1);
          const int hh = col >> 6,  dk = col & (DKH - 1);
          outB[((((size_t)bi * NH + hh) << 10) + s) * DKH + dk] = (__bf16)v;
        } else if constexpr (MODE == EP_GELU) {
          outB[(size_t)row * N + col] = (__bf16)gelu_f(v);
        } else { // EP_RES
          const size_t idx = (size_t)row * N + col;
          outF[idx] = v + res[idx];
        }
      }
    }
  }
}

// ---------------- flash attention (WMMA, online softmax) ----------------
__global__ __launch_bounds__(256)
void flash_attn_wmma(const __bf16* __restrict__ Q, const __bf16* __restrict__ Kb,
                     const __bf16* __restrict__ Vb, const int* __restrict__ mask,
                     __bf16* __restrict__ O) {
  __shared__ __bf16 sK[32 * 72];       // [kv][d]
  __shared__ __bf16 sVt[64 * 40];      // [d][kv]  (V transposed)
  __shared__ __bf16 sP[8][16 * 40];    // per-wave P scratch (C-layout -> A-layout relay)

  const int tid = threadIdx.x, lane = tid & 31, wave = tid >> 5;
  const int b = blockIdx.z, h = blockIdx.y, q0 = blockIdx.x * 128;
  const int qbase = q0 + wave * 16;
  const size_t bh = ((size_t)b * NH + h) * SEQ;

  const int r = lane & 15, kh = lane >> 4;

  // Q fragments live in registers for the whole kernel (A layout, K=d)
  const __bf16* qrow = Q + (bh + qbase + r) * DKH;
  const v16bf aq0 = cat8(*(const v8bf*)(qrow + kh * 8),      *(const v8bf*)(qrow + 16 + kh * 8));
  const v16bf aq1 = cat8(*(const v8bf*)(qrow + 32 + kh * 8), *(const v8bf*)(qrow + 48 + kh * 8));

  v8f zero = {};
  v8f oacc[4];
  #pragma unroll
  for (int n = 0; n < 4; ++n) oacc[n] = zero;
  float m_run[8], l_run[8];
  #pragma unroll
  for (int j = 0; j < 8; ++j) { m_run[j] = -3.0e38f; l_run[j] = 0.f; }

  const int lkv = tid >> 3;           // 0..31
  const int ld  = (tid & 7) << 3;     // 0,8,...,56
  const int* mbase = mask + ((size_t)b * SEQ + qbase) * SEQ;

  for (int t = 0; t < SEQ / 32; ++t) {
    const int kv0 = t * 32;
    __syncthreads();  // previous iteration's reads of sK/sVt complete
    // K tile via async LDS copy (no VGPR round-trip)
    async_ld16((unsigned)(size_t)&sK[lkv * 72 + ld],
               &Kb[(bh + kv0 + lkv) * DKH + ld]);
    // V tile: load + transpose into [d][kv]
    v8bf vv = *(const v8bf*)&Vb[(bh + kv0 + lkv) * DKH + ld];
    #pragma unroll
    for (int j = 0; j < 8; ++j) sVt[(ld + j) * 40 + lkv] = vv[j];
    wait_async0();
    __syncthreads();

    // scores: 16 q-rows x 32 kv (two 16x16 tiles), K-dim = d (64 -> 2 WMMA each)
    v8f sc[2];
    #pragma unroll
    for (int sn = 0; sn < 2; ++sn) {
      v8f a = zero;
      a = wmma_bf16(aq0, frag_b(&sK[(sn * 16) * 72], 72, 0,  lane), a);
      a = wmma_bf16(aq1, frag_b(&sK[(sn * 16) * 72], 72, 32, lane), a);
      sc[sn] = a;
    }

    // scale + mask + online softmax
    #pragma unroll
    for (int j = 0; j < 8; ++j) {
      const int qoff = (j + kh * 8) * SEQ + kv0;
      float s0 = sc[0][j] * 0.125f, s1 = sc[1][j] * 0.125f;
      s0 = mbase[qoff + r]      ? s0 : -1e9f;
      s1 = mbase[qoff + 16 + r] ? s1 : -1e9f;
      const float rm = rmax16(fmaxf(s0, s1));
      const float mn = fmaxf(m_run[j], rm);
      const float alpha = __expf(m_run[j] - mn);
      m_run[j] = mn;
      const float p0 = __expf(s0 - mn), p1 = __expf(s1 - mn);
      sc[0][j] = p0; sc[1][j] = p1;
      l_run[j] = l_run[j] * alpha + rsum16(p0 + p1);
      #pragma unroll
      for (int n = 0; n < 4; ++n) oacc[n][j] *= alpha;
    }

    // P: C-layout regs -> per-wave LDS -> A-layout fragment
    __bf16* pw = &sP[wave][0];
    #pragma unroll
    for (int j = 0; j < 8; ++j) {
      pw[(j + kh * 8) * 40 + r]      = (__bf16)sc[0][j];
      pw[(j + kh * 8) * 40 + 16 + r] = (__bf16)sc[1][j];
    }
    asm volatile("s_wait_dscnt 0x0" ::: "memory");
    const v16bf ap = frag_a(pw, 40, lane);

    // O += P @ V  (K-dim = 32 kv, columns = d)
    #pragma unroll
    for (int n = 0; n < 4; ++n)
      oacc[n] = wmma_bf16(ap, frag_b(&sVt[(n * 16) * 40], 40, 0, lane), oacc[n]);
  }

  // normalize + write [B,S,H*DK] bf16
  #pragma unroll
  for (int j = 0; j < 8; ++j) {
    const float inv = 1.f / l_run[j];
    const int row = qbase + j + kh * 8;
    #pragma unroll
    for (int n = 0; n < 4; ++n)
      O[((size_t)b * SEQ + row) * DIM + h * DKH + n * 16 + r] = (__bf16)(oacc[n][j] * inv);
  }
}

// ---------------- host orchestration ----------------
extern "C" void kernel_launch(void* const* d_in, const int* in_sizes, int n_in,
                              void* d_out, int out_size, void* d_ws, size_t ws_size,
                              hipStream_t stream) {
  (void)in_sizes; (void)n_in; (void)out_size; (void)ws_size;
  const float* x    = (const float*)d_in[0];
  const int*   mask = (const int*)d_in[1];
  const float* wq = (const float*)d_in[2];  const float* bq = (const float*)d_in[3];
  const float* wk = (const float*)d_in[4];  const float* bk = (const float*)d_in[5];
  const float* wv = (const float*)d_in[6];  const float* bv = (const float*)d_in[7];
  const float* wo = (const float*)d_in[8];  const float* bo = (const float*)d_in[9];
  const float* w1 = (const float*)d_in[10]; const float* b1 = (const float*)d_in[11];
  const float* w2 = (const float*)d_in[12]; const float* b2 = (const float*)d_in[13];
  const float* g1 = (const float*)d_in[14]; const float* be1 = (const float*)d_in[15];
  const float* g2 = (const float*)d_in[16]; const float* be2 = (const float*)d_in[17];

  const size_t BS = (size_t)BQ * SEQ;
  char* ws = (char*)d_ws;
  size_t off = 0;
  auto carve = [&](size_t bytes) -> char* {
    char* p = ws + off;
    off += (bytes + 255) & ~(size_t)255;
    return p;
  };
  __bf16* wq_b  = (__bf16*)carve((size_t)DIM * DIM * 2);
  __bf16* wk_b  = (__bf16*)carve((size_t)DIM * DIM * 2);
  __bf16* wv_b  = (__bf16*)carve((size_t)DIM * DIM * 2);
  __bf16* wo_b  = (__bf16*)carve((size_t)DIM * DIM * 2);
  __bf16* w1_b  = (__bf16*)carve((size_t)DFFN * DIM * 2);
  __bf16* w2_b  = (__bf16*)carve((size_t)DIM * DFFN * 2);
  __bf16* xn_b  = (__bf16*)carve(BS * DIM * 2);
  __bf16* q_b   = (__bf16*)carve(BS * DIM * 2);
  __bf16* k_b   = (__bf16*)carve(BS * DIM * 2);
  __bf16* v_b   = (__bf16*)carve(BS * DIM * 2);
  __bf16* at_b  = (__bf16*)carve(BS * DIM * 2);
  float*  x1    = (float*) carve(BS * DIM * 4);
  __bf16* xn2_b = (__bf16*)carve(BS * DIM * 2);
  __bf16* ff1_b = (__bf16*)carve(BS * DFFN * 2);

  // 1) weights -> bf16
  cvt_f32_bf16<<<2048, 256, 0, stream>>>(wq, wq_b, DIM * DIM);
  cvt_f32_bf16<<<2048, 256, 0, stream>>>(wk, wk_b, DIM * DIM);
  cvt_f32_bf16<<<2048, 256, 0, stream>>>(wv, wv_b, DIM * DIM);
  cvt_f32_bf16<<<2048, 256, 0, stream>>>(wo, wo_b, DIM * DIM);
  cvt_f32_bf16<<<2048, 256, 0, stream>>>(w1, w1_b, DFFN * DIM);
  cvt_f32_bf16<<<2048, 256, 0, stream>>>(w2, w2_b, DIM * DFFN);

  // 2) LN1 -> bf16
  layernorm_to_bf16<<<(int)BS, 256, 0, stream>>>(x, g1, be1, xn_b);

  // 3) Q,K,V projections (write head-interleaved [B,H,S,DK])
  dim3 gD(DIM / 128, (int)(BS / 128));
  gemm_bf16_wmma<EP_HEADS><<<gD, 256, 0, stream>>>(xn_b, wq_b, bq, nullptr, nullptr, q_b,
                                                   (int)BS, DIM, DIM);
  gemm_bf16_wmma<EP_HEADS><<<gD, 256, 0, stream>>>(xn_b, wk_b, bk, nullptr, nullptr, k_b,
                                                   (int)BS, DIM, DIM);
  gemm_bf16_wmma<EP_HEADS><<<gD, 256, 0, stream>>>(xn_b, wv_b, bv, nullptr, nullptr, v_b,
                                                   (int)BS, DIM, DIM);

  // 4) attention
  flash_attn_wmma<<<dim3(SEQ / 128, NH, BQ), 256, 0, stream>>>(q_b, k_b, v_b, mask, at_b);

  // 5) output projection + residual: x1 = x + attn @ wo^T + bo
  gemm_bf16_wmma<EP_RES><<<gD, 256, 0, stream>>>(at_b, wo_b, bo, x, x1, nullptr,
                                                 (int)BS, DIM, DIM);

  // 6) LN2 -> bf16
  layernorm_to_bf16<<<(int)BS, 256, 0, stream>>>(x1, g2, be2, xn2_b);

  // 7) FFN up + GELU
  dim3 gF(DFFN / 128, (int)(BS / 128));
  gemm_bf16_wmma<EP_GELU><<<gF, 256, 0, stream>>>(xn2_b, w1_b, b1, nullptr, nullptr, ff1_b,
                                                  (int)BS, DFFN, DIM);

  // 8) FFN down + residual -> final f32 output
  gemm_bf16_wmma<EP_RES><<<gD, 256, 0, stream>>>(ff1_b, w2_b, b2, x1, (float*)d_out, nullptr,
                                                 (int)BS, DIM, DFFN);
}
